// RNN_EI_13460427506020
// MI455X (gfx1250) — matrix operand order repaired and checked by reading, not compiled, not run on hardware
//
#include <hip/hip_runtime.h>

typedef __attribute__((ext_vector_type(16))) __bf16 v16bf;
typedef __attribute__((ext_vector_type(8)))  float  v8f;
typedef unsigned int uint32x4 __attribute__((ext_vector_type(4)));
typedef unsigned int uint32x8 __attribute__((ext_vector_type(8)));

#define B_SZ   128
#define IN_SZ  1024
#define N_SZ   4096
#define T_SZ   48
#define DECAY  0.9f
#define TILE_N 32
#define PAD    40                     // LDS row stride in bf16 (80B: 16B-aligned, conflict-free)
#define CHUNK_ELEMS (TILE_N * PAD)    // one staged B chunk (32 rows x 32 k, padded)
#define CHUNK_BYTES (CHUNK_ELEMS * 2)

// ---------------- elementwise conversion kernels ----------------

__global__ void cvt_f32_to_bf16(const float* __restrict__ in, __bf16* __restrict__ out, int n) {
  int i = blockIdx.x * blockDim.x + threadIdx.x;
  if (i < n) out[i] = (__bf16)in[i];
}

// in: R x C row-major f32  ->  out: C x R row-major bf16 (transposed)
__global__ void transpose_f32_to_bf16(const float* __restrict__ in, __bf16* __restrict__ out,
                                      int R, int C) {
  int i = blockIdx.x * blockDim.x + threadIdx.x;
  if (i < R * C) {
    int r = i / C;
    int c = i - r * C;
    out[(size_t)c * R + r] = (__bf16)in[i];
  }
}

// ---------------- fused recurrent step (elementwise) ----------------
__global__ void step_update(const float* __restrict__ drive,
                            float* __restrict__ xm,
                            const float* __restrict__ r,
                            float* __restrict__ out_f,
                            float* __restrict__ out_u,
                            __bf16* __restrict__ f_bf16,
                            int t) {
  int idx = blockIdx.x * blockDim.x + threadIdx.x;   // [0, B*N)
  int b = idx >> 12;               // / 4096
  int n = idx & (N_SZ - 1);
  float v = DECAY * xm[idx] + drive[idx] + r[idx];
  xm[idx] = v;
  float f = fmaxf(v, 0.0f);
  size_t o = ((size_t)b * T_SZ + t) * N_SZ + n;
  out_u[o] = v;
  out_f[o] = f;
  f_bf16[idx] = (__bf16)f;
}

// ---------------- Tensor Data Mover: 2D bf16 tile -> LDS ----------------
// Issues TENSOR_LOAD_TO_LDS for a (tile_k=32 x tile_rows=32) bf16 tile whose
// rows are K elements apart in global memory, landing in LDS with a 16B pad
// after every 64B row (pad_interval=3 -> every 16 DWORDs, pad_amount=3 -> 4
// DWORDs) so the LDS row stride is 80B == PAD bf16.  Tracked by TENSORcnt.
//
// ldsPtr is passed BOTH as descriptor payload (low 32 bits of the flat
// address == LDS byte address, per ISA aperture rules) AND as an asm input
// operand so the shared buffer escapes: without it LLVM proves Bs is never
// written and deletes the consumer ds_loads (observed in round 2).
__device__ __forceinline__ void tdm_load_b_chunk(const __bf16* gptr, const __bf16* ldsPtr,
                                                 int K, int nrows) {
  unsigned long long ga = (unsigned long long)(size_t)gptr;
  unsigned lds_off = (unsigned)(size_t)(const void*)ldsPtr;  // flat[31:0] == LDS offset
  uint32x4 g0;
  g0[0] = 1u;                                               // count=1, is_restore=0, no gather
  g0[1] = lds_off;                                          // LDS byte address of tile start
  g0[2] = (unsigned)(ga & 0xffffffffu);                     // global_addr[31:0]
  g0[3] = (unsigned)((ga >> 32) & 0x01ffffffu)              // global_addr[56:32]
        | 0x80000000u;                                      // type=2 ("image")
  uint32x8 g1;
  g1[0] = (1u << 16)                                        // data_size=1 -> 2 bytes
        | (1u << 20)                                        // pad_enable
        | (3u << 22)                                        // pad_interval: every 16 DWORDs
        | (3u << 25);                                       // pad_amount: 4 DWORDs
  g1[1] = ((unsigned)K & 0xffffu) << 16;                    // tensor_dim0[15:0]
  g1[2] = (((unsigned)K >> 16) & 0xffffu)                   // tensor_dim0[31:16]
        | (((unsigned)nrows & 0xffffu) << 16);              // tensor_dim1[15:0]
  g1[3] = (((unsigned)nrows >> 16) & 0xffffu)               // tensor_dim1[31:16]
        | (32u << 16);                                      // tile_dim0 = 32 (k)
  g1[4] = 32u;                                              // tile_dim1 = 32 rows, tile_dim2 = 0
  g1[5] = (unsigned)K;                                      // tensor_dim0_stride[31:0]
  g1[6] = 0u;                                               // stride0 hi / stride1 lo
  g1[7] = 0u;
  asm volatile("tensor_load_to_lds %0, %1"
               :
               : "s"(g0), "s"(g1), "s"(ldsPtr)              // ldsPtr: escape for alias analysis
               : "memory");
}

// ---------------- bf16 WMMA GEMM ----------------
// C[128, N] = A[128, K] @ Bt[N, K]^T + bias  (bias may be null)
// 256 threads = 8 wave32; wave w covers rows [16w,16w+16); block covers a
// TILE_N=32 column strip (two 16x16 C tiles per wave).  B chunks are DMA'd
// into double-buffered LDS by the Tensor Data Mover while WMMAs run.
__global__ __launch_bounds__(256, 2)
void gemm_bf16(const __bf16* __restrict__ A,
               const __bf16* __restrict__ Bt,
               const float* __restrict__ bias,
               float* __restrict__ C,
               int K, int N) {
  __shared__ __bf16 Bs[2 * CHUNK_ELEMS];

  const int wave = threadIdx.x >> 5;
  const int lane = threadIdx.x & 31;
  const int hi   = lane >> 4;              // lane half selects K sub-range (ISA 7.12.2)
  const int lo   = lane & 15;
  const int rowStart = wave * 16;
  const int colStart = blockIdx.x * TILE_N;

  v8f acc0 = {};
  v8f acc1 = {};

  const __bf16* aRow  = A + (size_t)(rowStart + lo) * K;
  const __bf16* bBase = Bt + (size_t)colStart * K;   // origin of this block's column strip

  // prologue: DMA chunk 0 into buffer 0
  if (wave == 0) tdm_load_b_chunk(bBase, &Bs[0], K, TILE_N);

  int ib = 0;
  for (int k0 = 0; k0 < K; k0 += 32) {
    if (wave == 0) __builtin_amdgcn_s_wait_tensorcnt(0);   // chunk k landed in LDS
    __syncthreads();

    // overlap: DMA chunk k+1 into the other buffer while we compute chunk k
    if (wave == 0 && k0 + 32 < K)
      tdm_load_b_chunk(bBase + (k0 + 32), &Bs[(ib ^ 1) * CHUNK_ELEMS], K, TILE_N);

    // A fragment, 16x32 bf16 strip (contiguous K order, matches B):
    //   lane half 0: K = k0+[0..15]; lane half 1: K = k0+16+[0..15]
    v16bf a;
    const __bf16* ap = aRow + k0 + hi * 16;
    ((uint4*)&a)[0] = *(const uint4*)(ap);        // elements 0..7
    ((uint4*)&a)[1] = *(const uint4*)(ap + 8);    // elements 8..15
    if (k0 + 32 < K) __builtin_prefetch(ap + 32, 0, 1);    // global_prefetch_b8

    const __bf16* buf = &Bs[ib * CHUNK_ELEMS];
    // B fragments (contiguous K order, matches A): lane half h holds 16 contiguous K, col = lo
    {
      v16bf b;
      const __bf16* bp = buf + (0 * 16 + lo) * PAD + hi * 16;
      ((uint4*)&b)[0] = ((const uint4*)bp)[0];
      ((uint4*)&b)[1] = ((const uint4*)bp)[1];
      acc0 = __builtin_amdgcn_wmma_f32_16x16x32_bf16(false, a, false, b,
                                                     (short)0, acc0, false, false);
    }
    {
      v16bf b;
      const __bf16* bp = buf + (1 * 16 + lo) * PAD + hi * 16;
      ((uint4*)&b)[0] = ((const uint4*)bp)[0];
      ((uint4*)&b)[1] = ((const uint4*)bp)[1];
      acc1 = __builtin_amdgcn_wmma_f32_16x16x32_bf16(false, a, false, b,
                                                     (short)0, acc1, false, false);
    }
    __syncthreads();   // all reads of buf[ib] done before it is re-targeted by the DMA
    ib ^= 1;
  }

  // C/D layout: VGPR v -> row = rowStart + 8*hi + v, col = colStart + 16*j + lo
  const int r0 = rowStart + hi * 8;
#pragma unroll
  for (int j = 0; j < 2; ++j) {
    const int col = colStart + j * 16 + lo;
    const float bv = bias ? bias[col] : 0.0f;
    v8f accr = j ? acc1 : acc0;
    float* cp = C + (size_t)r0 * N + col;
#pragma unroll
    for (int v = 0; v < 8; ++v)
      cp[(size_t)v * N] = accr[v] + bv;
  }
}

// ---------------- launch ----------------

extern "C" void kernel_launch(void* const* d_in, const int* in_sizes, int n_in,
                              void* d_out, int out_size, void* d_ws, size_t ws_size,
                              hipStream_t stream) {
  const float* x  = (const float*)d_in[0];   // [B, IN]
  const float* Wi = (const float*)d_in[1];   // [IN, N]
  const float* b0 = (const float*)d_in[2];   // [N]
  const float* Wr = (const float*)d_in[3];   // [N, N]

  float* out_f = (float*)d_out;                              // output [B,T,N]
  float* out_u = out_f + (size_t)B_SZ * T_SZ * N_SZ;         // act    [B,T,N]

  char* p = (char*)d_ws;
  __bf16* WiT = (__bf16*)p; p += (size_t)IN_SZ * N_SZ * sizeof(__bf16);   // [N, IN]
  __bf16* WrT = (__bf16*)p; p += (size_t)N_SZ * N_SZ * sizeof(__bf16);    // [N, N]
  __bf16* xb  = (__bf16*)p; p += (size_t)B_SZ * IN_SZ * sizeof(__bf16);   // [B, IN]
  __bf16* fb  = (__bf16*)p; p += (size_t)B_SZ * N_SZ * sizeof(__bf16);    // [B, N]
  float* drive = (float*)p; p += (size_t)B_SZ * N_SZ * sizeof(float);
  float* xm    = (float*)p; p += (size_t)B_SZ * N_SZ * sizeof(float);
  float* rbuf  = (float*)p; p += (size_t)B_SZ * N_SZ * sizeof(float);

  // one-time (per call) precision/layout conversion
  cvt_f32_to_bf16<<<(B_SZ * IN_SZ + 255) / 256, 256, 0, stream>>>(x, xb, B_SZ * IN_SZ);
  transpose_f32_to_bf16<<<(IN_SZ * N_SZ + 255) / 256, 256, 0, stream>>>(Wi, WiT, IN_SZ, N_SZ);
  transpose_f32_to_bf16<<<(unsigned)(((size_t)N_SZ * N_SZ + 255) / 256), 256, 0, stream>>>(
      Wr, WrT, N_SZ, N_SZ);

  hipMemsetAsync(xm,   0, (size_t)B_SZ * N_SZ * sizeof(float), stream);
  hipMemsetAsync(rbuf, 0, (size_t)B_SZ * N_SZ * sizeof(float), stream);

  // drive = x @ Wi + b0
  gemm_bf16<<<N_SZ / TILE_N, 256, 0, stream>>>(xb, WiT, b0, drive, IN_SZ, N_SZ);

  // serial scan: each step depends on r from the previous step's GEMM
  for (int t = 0; t < T_SZ; ++t) {
    step_update<<<(B_SZ * N_SZ) / 256, 256, 0, stream>>>(drive, xm, rbuf, out_f, out_u, fb, t);
    if (t + 1 < T_SZ)   // last r is unused
      gemm_bf16<<<N_SZ / TILE_N, 256, 0, stream>>>(fb, WrT, nullptr, rbuf, N_SZ, N_SZ);
  }
}
